// MoELayer_50697793962046
// MI455X (gfx1250) — compile-verified, matrix operand order checked by reading, not run on hardware
//
#include <hip/hip_runtime.h>
#include <hip/hip_bf16.h>
#include <math.h>

// ---------------- problem constants (match reference) ----------------
#define TT   2048   // tokens (B*S)
#define DM   1024   // d_model
#define EM   8      // experts
#define FM   4096   // d_ff
#define TOPK 2

typedef __bf16 bf16;
typedef __attribute__((ext_vector_type(2)))  __bf16 bf16x2;
typedef __attribute__((ext_vector_type(16))) __bf16 v16bf;
typedef __attribute__((ext_vector_type(8)))  float  v8f;

// ---------------- kernel 0: zero the tiny counters ----------------
__global__ void moe_init(int* counts, int* offsets) {
    int i = threadIdx.x;
    if (i < EM) { counts[i] = 0; offsets[i] = 0; }
}

// ---------------- kernel 1: router (logits, top-2, softmax) ----------------
__global__ void moe_router(const float* __restrict__ x, const float* __restrict__ Wr,
                           int* __restrict__ counts,
                           int* __restrict__ topk_e, float* __restrict__ gate) {
    int t = blockIdx.x * blockDim.x + threadIdx.x;
    if (t >= TT) return;
    const float* xr = x + (size_t)t * DM;
    float acc[EM];
#pragma unroll
    for (int e = 0; e < EM; ++e) acc[e] = 0.f;
    for (int d = 0; d < DM; ++d) {
        float xv = xr[d];
        const float* wr = Wr + (size_t)d * EM;
#pragma unroll
        for (int e = 0; e < EM; ++e) acc[e] += xv * wr[e];
    }
    // top-2 (prefer lower index on ties, like jax top_k)
    int e0 = 0; float v0 = acc[0];
#pragma unroll
    for (int e = 1; e < EM; ++e) if (acc[e] > v0) { v0 = acc[e]; e0 = e; }
    int e1 = -1; float v1 = -3.4e38f;
#pragma unroll
    for (int e = 0; e < EM; ++e) if (e != e0 && acc[e] > v1) { v1 = acc[e]; e1 = e; }
    float p0 = 1.f / (1.f + __expf(v1 - v0));   // softmax over {v0, v1}
    float p1 = 1.f - p0;
    topk_e[t * TOPK + 0] = e0;
    topk_e[t * TOPK + 1] = e1;
    gate[t * TOPK + 0] = p0;
    gate[t * TOPK + 1] = p1;
    atomicAdd(&counts[e0], 1);
    atomicAdd(&counts[e1], 1);
}

// ---------------- kernel 2: exclusive scan over 8 expert counts ----------------
__global__ void moe_scan(const int* __restrict__ counts, int* __restrict__ bases) {
    if (threadIdx.x == 0) {
        int s = 0;
#pragma unroll
        for (int e = 0; e < EM; ++e) { bases[e] = s; s += counts[e]; }
    }
}

// ---------------- kernel 3: build compacted slot lists ----------------
__global__ void moe_fill(const int* __restrict__ topk_e, const float* __restrict__ gate,
                         const int* __restrict__ bases, int* __restrict__ offsets,
                         int* __restrict__ slot_token, float* __restrict__ slot_gate,
                         int* __restrict__ tk_slot) {
    int i = blockIdx.x * blockDim.x + threadIdx.x;   // i = t*TOPK + k
    if (i >= TT * TOPK) return;
    int t = i / TOPK;
    int e = topk_e[i];
    int slot = bases[e] + atomicAdd(&offsets[e], 1);
    slot_token[slot] = t;
    slot_gate[slot]  = gate[i];
    tk_slot[i]       = slot;
}

// ---------------- kernel 4: cast x to bf16 ----------------
__global__ void moe_xcast(const float* __restrict__ x, bf16* __restrict__ xb) {
    int i = blockIdx.x * blockDim.x + threadIdx.x;
    if (i < TT * DM) xb[i] = (bf16)x[i];
}

// --------- WMMA fragment index helpers (ISA 7.12.2, wave32) ---------
// A (16x32, 16-bit): lane = row (mod 16); element i -> K = kb + (i<8 ? i : i+8),
//   kb = (lane>=16 ? 8 : 0).
// B (32x16, 16-bit): lane = col (mod 16); same K striping per lane half.
// C/D (16x16 f32):   lane = N (mod 16); VGPR i -> M = i + (lane>=16 ? 8 : 0).

__device__ __forceinline__ float fast_silu(float h) {
    float sig = __builtin_amdgcn_rcpf(1.f + __expf(-h));
    return h * sig;
}

// ---------------- kernel 5: GEMM1  h = x@W1, g = x@W3, act = silu(h)*g ----------------
// Block tile: M=64 slots x N=64 f-columns, 128 threads (4 waves); wave w owns 16 cols.
// grid: (FM/64, EM * (TT/64))
__global__ void __launch_bounds__(128)
moe_gemm1(const bf16* __restrict__ xb,
          const float* __restrict__ W1, const float* __restrict__ W3,
          const int* __restrict__ bases, const int* __restrict__ counts,
          const int* __restrict__ slot_token, bf16* __restrict__ act) {
    const int MTILES = TT / 64;                 // 32 worst-case M tiles / expert
    int e  = blockIdx.y / MTILES;
    int mt = blockIdx.y % MTILES;
    int cnt = counts[e];
    if (mt * 64 >= cnt) return;                 // no rows for this tile
    int base = bases[e];
    int row0 = base + mt * 64;
    int rows = min(64, cnt - mt * 64);

    int tid  = threadIdx.x;
    int wave = tid >> 5;
    int lane = tid & 31;
    int fbase = blockIdx.x * 64;
    int fw    = fbase + wave * 16;              // this wave's 16 output columns

    __shared__ bf16 Atile[64][34];              // padded vs. bank conflicts
    __shared__ bf16 B1t[32][72];
    __shared__ bf16 B3t[32][72];
    __shared__ int  tok[64];
    if (tid < 64) {
        int r = tid;
        tok[r] = slot_token[row0 + (r < rows ? r : 0)];
    }
    __syncthreads();

    const float* w1e = W1 + (size_t)e * DM * FM;
    const float* w3e = W3 + (size_t)e * DM * FM;

    int rowA = lane & 15;
    int colB = (lane & 15) + wave * 16;
    int kb   = (lane >> 4) * 8;

    v8f acc_h[4] = {v8f{}, v8f{}, v8f{}, v8f{}};
    v8f acc_g[4] = {v8f{}, v8f{}, v8f{}, v8f{}};

    // staging index maps (fixed per thread)
    int ar = tid >> 1;                // A: 0..63 rows
    int ac = (tid & 1) * 16;          // A: 2 halves of 32 cols
    int kr = tid >> 2;                // B: 0..31 k-rows
    int fc = (tid & 3) * 16;          // B: 4 chunks of 16 cols

    for (int kc = 0; kc < DM; kc += 32) {
        // ---- stage A: 64x32 bf16 (gathered tokens), 16 bf16 / thread ----
        {
            const unsigned int* src =
                (const unsigned int*)(xb + (size_t)tok[ar] * DM + kc + ac);
            unsigned int* dst = (unsigned int*)&Atile[ar][ac];
#pragma unroll
            for (int j = 0; j < 8; ++j) dst[j] = src[j];
        }
        // ---- stage B1/B3: 32x64 fp32 -> bf16, coalesced float4 loads ----
        {
            const float4* s1 = (const float4*)(w1e + (size_t)(kc + kr) * FM + fbase + fc);
            const float4* s3 = (const float4*)(w3e + (size_t)(kc + kr) * FM + fbase + fc);
            bf16x2* d1 = (bf16x2*)&B1t[kr][fc];
            bf16x2* d3 = (bf16x2*)&B3t[kr][fc];
#pragma unroll
            for (int j = 0; j < 4; ++j) {
                float4 v1 = s1[j];
                float4 v3 = s3[j];
                d1[2 * j + 0] = bf16x2{(bf16)v1.x, (bf16)v1.y};
                d1[2 * j + 1] = bf16x2{(bf16)v1.z, (bf16)v1.w};
                d3[2 * j + 0] = bf16x2{(bf16)v3.x, (bf16)v3.y};
                d3[2 * j + 1] = bf16x2{(bf16)v3.z, (bf16)v3.w};
            }
        }
        if (kc + 32 < DM) {   // prefetch next K-step weight rows
            __builtin_prefetch(w1e + (size_t)(kc + 32 + kr) * FM + fbase + fc, 0, 0);
            __builtin_prefetch(w3e + (size_t)(kc + 32 + kr) * FM + fbase + fc, 0, 0);
        }
        __syncthreads();

        // ---- B fragments for this wave's 16 columns ----
        v16bf b1, b3;
#pragma unroll
        for (int i = 0; i < 16; ++i) {
            int k = kb + (i < 8 ? i : i + 8);
            b1[i] = B1t[k][colB];
            b3[i] = B3t[k][colB];
        }
        // ---- 4 M-subtiles, 8 WMMAs per wave per K-step ----
#pragma unroll
        for (int m = 0; m < 4; ++m) {
            v16bf a;
#pragma unroll
            for (int i = 0; i < 16; ++i) {
                int k = kb + (i < 8 ? i : i + 8);
                a[i] = Atile[m * 16 + rowA][k];
            }
            acc_h[m] = __builtin_amdgcn_wmma_f32_16x16x32_bf16(false, a, false, b1,
                                                               (short)0, acc_h[m], false, false);
            acc_g[m] = __builtin_amdgcn_wmma_f32_16x16x32_bf16(false, a, false, b3,
                                                               (short)0, acc_g[m], false, false);
        }
        __syncthreads();
    }

    // epilogue: act = silu(h) * g, bf16 out
    int Mb = (lane >> 4) * 8;
    int n  = lane & 15;
#pragma unroll
    for (int m = 0; m < 4; ++m) {
#pragma unroll
        for (int i = 0; i < 8; ++i) {
            int M = m * 16 + Mb + i;
            if (M < rows) {
                float h = acc_h[m][i], g = acc_g[m][i];
                act[(size_t)(row0 + M) * FM + fw + n] = (bf16)(fast_silu(h) * g);
            }
        }
    }
}

// ---------------- kernel 6: GEMM2  y = act @ W2, scaled by gate ----------------
// Block tile: M=64 slots x N=64 d-columns, 128 threads (4 waves).
// grid: (DM/64, EM * (TT/64))
__global__ void __launch_bounds__(128)
moe_gemm2(const bf16* __restrict__ act, const float* __restrict__ W2,
          const int* __restrict__ bases, const int* __restrict__ counts,
          const float* __restrict__ slot_gate, float* __restrict__ ypart) {
    const int MTILES = TT / 64;
    int e  = blockIdx.y / MTILES;
    int mt = blockIdx.y % MTILES;
    int cnt = counts[e];
    if (mt * 64 >= cnt) return;
    int base = bases[e];
    int row0 = base + mt * 64;
    int rows = min(64, cnt - mt * 64);

    int tid  = threadIdx.x;
    int wave = tid >> 5;
    int lane = tid & 31;
    int dbase = blockIdx.x * 64;
    int dw    = dbase + wave * 16;

    __shared__ bf16 Atile[64][34];
    __shared__ bf16 Bt[32][72];

    const float* w2e = W2 + (size_t)e * FM * DM;

    int rowA = lane & 15;
    int colB = (lane & 15) + wave * 16;
    int kb   = (lane >> 4) * 8;

    v8f acc[4] = {v8f{}, v8f{}, v8f{}, v8f{}};

    int ar = tid >> 1;
    int ac = (tid & 1) * 16;
    int kr = tid >> 2;
    int fc = (tid & 3) * 16;
    int asr = row0 + (ar < rows ? ar : 0);      // clamp to valid slot

    for (int kc = 0; kc < FM; kc += 32) {
        {
            const unsigned int* src =
                (const unsigned int*)(act + (size_t)asr * FM + kc + ac);
            unsigned int* dst = (unsigned int*)&Atile[ar][ac];
#pragma unroll
            for (int j = 0; j < 8; ++j) dst[j] = src[j];
        }
        {
            const float4* s2 = (const float4*)(w2e + (size_t)(kc + kr) * DM + dbase + fc);
            bf16x2* d2 = (bf16x2*)&Bt[kr][fc];
#pragma unroll
            for (int j = 0; j < 4; ++j) {
                float4 v = s2[j];
                d2[2 * j + 0] = bf16x2{(bf16)v.x, (bf16)v.y};
                d2[2 * j + 1] = bf16x2{(bf16)v.z, (bf16)v.w};
            }
        }
        if (kc + 32 < FM) {
            __builtin_prefetch(w2e + (size_t)(kc + 32 + kr) * DM + dbase + fc, 0, 0);
        }
        __syncthreads();

        v16bf b;
#pragma unroll
        for (int i = 0; i < 16; ++i) {
            int k = kb + (i < 8 ? i : i + 8);
            b[i] = Bt[k][colB];
        }
#pragma unroll
        for (int m = 0; m < 4; ++m) {
            v16bf a;
#pragma unroll
            for (int i = 0; i < 16; ++i) {
                int k = kb + (i < 8 ? i : i + 8);
                a[i] = Atile[m * 16 + rowA][k];
            }
            acc[m] = __builtin_amdgcn_wmma_f32_16x16x32_bf16(false, a, false, b,
                                                             (short)0, acc[m], false, false);
        }
        __syncthreads();
    }

    int Mb = (lane >> 4) * 8;
    int n  = lane & 15;
#pragma unroll
    for (int m = 0; m < 4; ++m) {
#pragma unroll
        for (int i = 0; i < 8; ++i) {
            int M = m * 16 + Mb + i;
            if (M < rows) {
                float gw = slot_gate[row0 + M];
                ypart[(size_t)(row0 + M) * DM + dw + n] = acc[m][i] * gw;
            }
        }
    }
}

// ---------------- kernel 7: combine the two expert slots per token ----------------
__global__ void moe_combine(const float* __restrict__ ypart, const int* __restrict__ tk_slot,
                            float* __restrict__ out) {
    int i = blockIdx.x * blockDim.x + threadIdx.x;
    if (i >= TT * DM) return;
    int t = i / DM;
    int d = i - t * DM;
    int s0 = tk_slot[t * TOPK + 0];
    int s1 = tk_slot[t * TOPK + 1];
    out[i] = ypart[(size_t)s0 * DM + d] + ypart[(size_t)s1 * DM + d];
}

// ---------------- host launcher ----------------
extern "C" void kernel_launch(void* const* d_in, const int* in_sizes, int n_in,
                              void* d_out, int out_size, void* d_ws, size_t ws_size,
                              hipStream_t stream) {
    (void)in_sizes; (void)n_in; (void)out_size; (void)ws_size;
    const float* x  = (const float*)d_in[0];
    const float* Wr = (const float*)d_in[1];
    const float* W1 = (const float*)d_in[2];
    const float* W2 = (const float*)d_in[3];
    const float* W3 = (const float*)d_in[4];
    float* out = (float*)d_out;

    // carve workspace (256B aligned regions)
    char* w = (char*)d_ws;
    auto carve = [&](size_t bytes) -> char* {
        char* p = w;
        w += (bytes + 255) & ~(size_t)255;
        return p;
    };
    int*   counts     = (int*)  carve(EM * sizeof(int));
    int*   offsets    = (int*)  carve(EM * sizeof(int));
    int*   bases      = (int*)  carve(EM * sizeof(int));
    int*   topk_e     = (int*)  carve((size_t)TT * TOPK * sizeof(int));
    float* gate       = (float*)carve((size_t)TT * TOPK * sizeof(float));
    int*   slot_token = (int*)  carve((size_t)TT * TOPK * sizeof(int));
    float* slot_gate  = (float*)carve((size_t)TT * TOPK * sizeof(float));
    int*   tk_slot    = (int*)  carve((size_t)TT * TOPK * sizeof(int));
    bf16*  xb         = (bf16*) carve((size_t)TT * DM * sizeof(bf16));
    bf16*  act        = (bf16*) carve((size_t)TT * TOPK * FM * sizeof(bf16));
    float* ypart      = (float*)carve((size_t)TT * TOPK * DM * sizeof(float));

    hipLaunchKernelGGL(moe_init, dim3(1), dim3(64), 0, stream, counts, offsets);
    hipLaunchKernelGGL(moe_router, dim3(TT / 256), dim3(256), 0, stream,
                       x, Wr, counts, topk_e, gate);
    hipLaunchKernelGGL(moe_scan, dim3(1), dim3(32), 0, stream, counts, bases);
    hipLaunchKernelGGL(moe_fill, dim3((TT * TOPK) / 256), dim3(256), 0, stream,
                       topk_e, gate, bases, offsets, slot_token, slot_gate, tk_slot);
    hipLaunchKernelGGL(moe_xcast, dim3((TT * DM) / 256), dim3(256), 0, stream, x, xb);

    dim3 g1(FM / 64, EM * (TT / 64));
    hipLaunchKernelGGL(moe_gemm1, g1, dim3(128), 0, stream,
                       xb, W1, W3, bases, counts, slot_token, act);

    dim3 g2(DM / 64, EM * (TT / 64));
    hipLaunchKernelGGL(moe_gemm2, g2, dim3(128), 0, stream,
                       act, W2, bases, counts, slot_gate, ypart);

    hipLaunchKernelGGL(moe_combine, dim3((TT * DM) / 256), dim3(256), 0, stream,
                       ypart, tk_slot, out);
}
